// TransPhormer_72808285602170
// MI455X (gfx1250) — matrix operand order repaired and checked by reading, not compiled, not run on hardware
//
#include <hip/hip_runtime.h>
#include <hip/hip_fp16.h>
#include <math.h>

typedef _Float16 f16;
typedef __attribute__((ext_vector_type(16))) _Float16 v16h;
typedef __attribute__((ext_vector_type(8)))  _Float16 v8h;
typedef __attribute__((ext_vector_type(8)))  float    v8f;

#define NN 10000
#define NE 160000
#define NTILES (NE/16)      // 10000
#define NW 2                // waves (tiles) per block in edge kernels

#define INV_SQRT3 0.5773502691896258f
#define INV_SQRT2 0.7071067811865476f
#define INV_SQRT32 0.17677669529663687f
#define INV_SQRT48 0.14433756729740643f
#define INV_SQRT128 0.08838834764831845f
#define ATTN_SCALE 0.14433756729740643f   // 1/sqrt(M0+M1)=1/sqrt(48)

// ---------------------------------------------------------------- helpers ---
__device__ inline v8f wmma_f16(v16h a, v16h b, v8f c) {
  return __builtin_amdgcn_wmma_f32_16x16x32_f16(false, a, false, b, (short)0, c,
                                                false, false);
}

// A tile in LDS: [16][64] f16 row-major (m, k).  Load the WMMA A fragment for
// k-chunk kk (0 or 32).  Lane layout (contiguous k order, same as PB):
//   elems 0..15 -> k = kk + half*16 + i
__device__ inline v16h load_A(const f16* At, int kk, int lane) {
  int m = lane & 15, half = lane >> 4;
  const f16* p = At + m * 64 + kk + half * 16;
  v8h lo = *(const v8h*)(p);
  v8h hi = *(const v8h*)(p + 8);
  v16h a;
#pragma unroll
  for (int i = 0; i < 8; ++i) { a[i] = lo[i]; a[8 + i] = hi[i]; }
  return a;
}

__device__ inline v16h load_B(const f16* PB, int frag, int lane) {
  return *(const v16h*)(PB + frag * 512 + lane * 16);
}

__device__ inline void atomicMaxFloat(float* addr, float v) {
  if (v >= 0.f) atomicMax((int*)addr, __float_as_int(v));
  else          atomicMin((unsigned int*)addr, __float_as_uint(v));
}

// ------------------------------------------------------------- prepack ------
// PB layout (f16): [key0: 32 frags][val0: 32 frags][key1: 16 frags][val1: 16]
// frag = 32 lanes * 16 halfs = 512 f16.  For frag (nb,kt): lane half*16+n,
// elem j holds B[kt*32 + half*16 + j][col], col = nofs + nb*16 + n, scaled.
__global__ void prepack_kernel(const float* __restrict__ Wkv0,
                               const float* __restrict__ Wkv1,
                               f16* __restrict__ PB) {
  int i = blockIdx.x * blockDim.x + threadIdx.x;   // 0 .. 49151
  int j = i & 15, lane = (i >> 4) & 31, f = i >> 9;
  const float* W; int ncols, nofs, fl, K; float scale;
  if (f < 32)      { W = Wkv0; ncols = 512; nofs = 0;   fl = f;      K = 48; scale = INV_SQRT48; }
  else if (f < 64) { W = Wkv0; ncols = 512; nofs = 256; fl = f - 32; K = 48; scale = INV_SQRT48; }
  else if (f < 80) { W = Wkv1; ncols = 256; nofs = 0;   fl = f - 64; K = 64; scale = 0.125f; }
  else             { W = Wkv1; ncols = 256; nofs = 128; fl = f - 80; K = 64; scale = 0.125f; }
  int nb = fl >> 1, kt = fl & 1;
  int half = lane >> 4, n = lane & 15;
  int k = kt * 32 + half * 16 + j;
  int col = nofs + nb * 16 + n;
  float v = (k < K) ? W[k * ncols + col] * scale : 0.f;
  PB[i] = (f16)v;
}

// ---------------------------------------------------------------- init ------
__global__ void init_kernel(float* segmax, float* segsum, float* msg) {
  int i = blockIdx.x * blockDim.x + threadIdx.x;
  if (i < NN * 8) { ((int*)segmax)[i] = 0xFF800000; segsum[i] = 0.f; }
  if (i < NN * 640) msg[i] = 0.f;
}

// ---------------------------------------------------------------- nodes -----
// One wave per node: layernorm/vecnorm then q (640), s0,s1,d0,d1 (160).
__global__ __launch_bounds__(256) void node_kernel(
    const float* __restrict__ node, const float* __restrict__ gamma0,
    const float* __restrict__ gamma1, const float* __restrict__ Wq0,
    const float* __restrict__ Wq1, const float* __restrict__ Wsrc0,
    const float* __restrict__ Wsrc1, const float* __restrict__ Wdst0,
    const float* __restrict__ Wdst1, float* __restrict__ q,
    float* __restrict__ sd) {
  int w = threadIdx.x >> 5, lane = threadIdx.x & 31;
  int nid = blockIdx.x * 8 + w;
  __shared__ float h0s[8][64];
  __shared__ float h1s[8][32][3];
  const float* nr = node + (size_t)nid * 160;
  float a = nr[lane], b = nr[lane + 32];
  float s = a + b;
#pragma unroll
  for (int m = 16; m; m >>= 1) s += __shfl_xor(s, m);
  float mu = s * (1.f / 64.f);
  float da = a - mu, db = b - mu;
  float vv = da * da + db * db;
#pragma unroll
  for (int m = 16; m; m >>= 1) vv += __shfl_xor(vv, m);
  float inv = rsqrtf(vv * (1.f / 64.f) + 1e-5f);
  h0s[w][lane]      = da * inv * gamma0[lane];
  h0s[w][lane + 32] = db * inv * gamma0[lane + 32];
  float x0 = nr[64 + lane * 3], x1 = nr[64 + lane * 3 + 1], x2 = nr[64 + lane * 3 + 2];
  float vn = x0 * x0 + x1 * x1 + x2 * x2;
#pragma unroll
  for (int m = 16; m; m >>= 1) vn += __shfl_xor(vn, m);
  float g = gamma1[lane] * rsqrtf(vn * (1.f / 32.f) + 1e-5f);
  h1s[w][lane][0] = x0 * g; h1s[w][lane][1] = x1 * g; h1s[w][lane][2] = x2 * g;
  __syncthreads();

  float* qrow = q + (size_t)nid * 640;
  for (int o = lane; o < 640; o += 32) {
    int h = o / 80, j = o % 80;
    float acc = 0.f;
    if (j < 32) {
      int col = h * 32 + j;
      for (int m = 0; m < 64; ++m) acc += h0s[w][m] * Wq0[m * 256 + col];
      acc *= 0.125f;
    } else {
      int jj = j - 32; int u = jj / 3, t = jj % 3; int col = h * 16 + u;
      for (int m = 0; m < 32; ++m) acc += h1s[w][m][t] * Wq1[m * 128 + col];
      acc *= INV_SQRT32;
    }
    qrow[o] = acc;
  }
  float* sdrow = sd + (size_t)nid * 160;
  for (int o = lane; o < 160; o += 32) {
    float acc = 0.f;
    if (o < 32) {
      for (int m = 0; m < 64; ++m) acc += h0s[w][m] * Wsrc0[m * 32 + o];
      acc *= 0.125f;
    } else if (o < 80) {
      int jj = o - 32; int u = jj / 3, t = jj % 3;
      for (int m = 0; m < 32; ++m) acc += h1s[w][m][t] * Wsrc1[m * 16 + u];
      acc *= INV_SQRT32;
    } else if (o < 112) {
      int c = o - 80;
      for (int m = 0; m < 64; ++m) acc += h0s[w][m] * Wdst0[m * 32 + c];
      acc *= 0.125f;
    } else {
      int jj = o - 112; int u = jj / 3, t = jj % 3;
      for (int m = 0; m < 32; ++m) acc += h1s[w][m][t] * Wdst1[m * 16 + u];
      acc *= INV_SQRT32;
    }
    sdrow[o] = acc;
  }
}

// --------------------------------------------------------- edge tile LDS ----
struct EdgeSmem {
  int   srcI[NW][16];
  int   dstI[NW][16];
  float y[NW][16][4];          // rsh
  float x0[NW][16][32];
  float x1[NW][16][16][3];
  float wt[NW][16][112];       // rbf @ W_rbf
  f16   A[NW][4][16][64];      // 0: cp0 (padded K 48->64); 1..3: cp1 comp t
  float lacc[NW][16][8];       // logits (kernel B) / attn weights (kernel D)
};

__device__ void build_tiles(int tile, int w, int lane,
                            const int* __restrict__ srcIdx,
                            const int* __restrict__ dstIdx,
                            const float* __restrict__ rbf,
                            const float* __restrict__ rsh,
                            const float* __restrict__ sd,
                            const float* __restrict__ Wrbf, EdgeSmem& S) {
  int e0 = tile * 16;
  if (lane < 16) {
    S.srcI[w][lane] = srcIdx[e0 + lane];
    S.dstI[w][lane] = dstIdx[e0 + lane];
  }
  for (int i = lane; i < 64; i += 32)
    S.y[w][i >> 2][i & 3] = rsh[(e0 + (i >> 2)) * 4 + (i & 3)];
  __syncthreads();
  for (int i = lane; i < 512; i += 32) {
    int m = i >> 5, c = i & 31;
    S.x0[w][m][c] = sd[(size_t)S.srcI[w][m] * 160 + c] +
                    sd[(size_t)S.dstI[w][m] * 160 + 80 + c];
  }
  for (int i = lane; i < 768; i += 32) {
    int m = i / 48, j = i % 48;
    S.x1[w][m][j / 3][j % 3] = sd[(size_t)S.srcI[w][m] * 160 + 32 + j] +
                               sd[(size_t)S.dstI[w][m] * 160 + 112 + j];
  }
  for (int i = lane; i < 1792; i += 32) {
    int m = i / 112, c = i % 112;
    const float* r = rbf + (size_t)(e0 + m) * 16;
    float acc = 0.f;
#pragma unroll
    for (int k = 0; k < 16; ++k) acc += r[k] * Wrbf[k * 112 + c];
    S.wt[w][m][c] = acc;
  }
  __syncthreads();
  // cp0 padded K 48->64
  for (int i = lane; i < 1024; i += 32) {
    int m = i >> 6, k = i & 63;
    float v = 0.f;
    if (k < 32) v = S.wt[w][m][k] * S.x0[w][m][k] * S.y[w][m][0];
    else if (k < 48) {
      int u = k - 32;
      v = S.wt[w][m][32 + u] *
          (S.x1[w][m][u][0] * S.y[w][m][1] + S.x1[w][m][u][1] * S.y[w][m][2] +
           S.x1[w][m][u][2] * S.y[w][m][3]) * INV_SQRT3;
    }
    S.A[w][0][m][k] = (f16)v;
  }
  // cp1, per component t
  for (int i = lane; i < 3072; i += 32) {
    int t = i >> 10, r = i & 1023, m = r >> 6, u = r & 63;
    float v;
    if (u < 32) v = S.wt[w][m][48 + u] * S.x0[w][m][u] * S.y[w][m][1 + t];
    else if (u < 48) {
      int uu = u - 32;
      v = S.wt[w][m][80 + uu] * S.x1[w][m][uu][t] * S.y[w][m][0];
    } else {
      int uu = u - 48, t1 = (t + 1) % 3, t2 = (t + 2) % 3;
      v = S.wt[w][m][96 + uu] *
          (S.x1[w][m][uu][t1] * S.y[w][m][1 + t2] -
           S.x1[w][m][uu][t2] * S.y[w][m][1 + t1]) * INV_SQRT2;
    }
    S.A[w][1 + t][m][u] = (f16)v;
  }
  __syncthreads();
}

// --------------------------------------------------------- edge key pass ----
// Per head h (8): 5 fragments (kv0 n-tiles 2h,2h+1; kv1 comps t=0..2 at nb=h).
// Per-lane partials accumulated across the 5 fragments, ONE butterfly per head.
__global__ __launch_bounds__(64) void edge_key_kernel(
    const int* __restrict__ srcIdx, const int* __restrict__ dstIdx,
    const float* __restrict__ rbf, const float* __restrict__ rsh,
    const float* __restrict__ sd, const float* __restrict__ Wrbf,
    const f16* __restrict__ PB, const float* __restrict__ q,
    float* __restrict__ logits, float* __restrict__ segmax) {
  __shared__ EdgeSmem S;
  int w = threadIdx.x >> 5, lane = threadIdx.x & 31;
  int tile = blockIdx.x * NW + w;
  build_tiles(tile, w, lane, srcIdx, dstIdx, rbf, rsh, sd, Wrbf, S);
  int half = lane >> 4, n = lane & 15;

  // hoist all A fragments (loop invariant)
  v16h A0k0 = load_A(&S.A[w][0][0][0], 0, lane);
  v16h A0k1 = load_A(&S.A[w][0][0][0], 32, lane);
  v16h A1k0[3], A1k1[3];
#pragma unroll
  for (int t = 0; t < 3; ++t) {
    A1k0[t] = load_A(&S.A[w][1 + t][0][0], 0, lane);
    A1k1[t] = load_A(&S.A[w][1 + t][0][0], 32, lane);
  }
  const f16* PB1 = PB + 32768;

  for (int h = 0; h < 8; ++h) {
    float p[8];
#pragma unroll
    for (int r = 0; r < 8; ++r) p[r] = 0.f;
    // kv0 key n-tiles nb = 2h, 2h+1
#pragma unroll
    for (int s = 0; s < 2; ++s) {
      int nb = 2 * h + s;
      v8f acc = {};
      acc = wmma_f16(A0k0, load_B(PB, nb * 2 + 0, lane), acc);
      acc = wmma_f16(A0k1, load_B(PB, nb * 2 + 1, lane), acc);
      int j = s * 16 + n;
#pragma unroll
      for (int r = 0; r < 8; ++r) {
        int m = half * 8 + r;
        p[r] += acc[r] * q[(size_t)S.srcI[w][m] * 640 + h * 80 + j];
      }
    }
    // kv1 key comps t = 0..2, n-tile nb = h
#pragma unroll
    for (int t = 0; t < 3; ++t) {
      v8f acc = {};
      acc = wmma_f16(A1k0[t], load_B(PB1, h * 2 + 0, lane), acc);
      acc = wmma_f16(A1k1[t], load_B(PB1, h * 2 + 1, lane), acc);
      int j = 32 + n * 3 + t;
#pragma unroll
      for (int r = 0; r < 8; ++r) {
        int m = half * 8 + r;
        p[r] += acc[r] * q[(size_t)S.srcI[w][m] * 640 + h * 80 + j];
      }
    }
    // one butterfly reduction across the 16 n-lanes per half
#pragma unroll
    for (int mask = 1; mask < 16; mask <<= 1)
#pragma unroll
      for (int r = 0; r < 8; ++r) p[r] += __shfl_xor(p[r], mask);
    if (n == 0)
#pragma unroll
      for (int r = 0; r < 8; ++r) S.lacc[w][half * 8 + r][h] = p[r];
  }
  __syncthreads();
  for (int i = lane; i < 128; i += 32) {
    int m = i >> 3, h = i & 7;
    float lg = S.lacc[w][m][h] * ATTN_SCALE;
    logits[(size_t)(tile * 16 + m) * 8 + h] = lg;
    atomicMaxFloat(&segmax[(size_t)S.dstI[w][m] * 8 + h], lg);
  }
}

// ---------------------------------------------------------- softmax mid -----
__global__ void softmax_kernel(const int* __restrict__ dstIdx,
                               const float* __restrict__ segmax,
                               float* __restrict__ logits,
                               float* __restrict__ segsum) {
  int i = blockIdx.x * blockDim.x + threadIdx.x;
  if (i >= NE * 8) return;
  int e = i >> 3, h = i & 7;
  int d = dstIdx[e];
  float a = __expf(logits[i] - segmax[d * 8 + h]);
  logits[i] = a;
  atomicAdd(&segsum[d * 8 + h], a);
}

// --------------------------------------------------------- edge val pass ----
__global__ __launch_bounds__(64) void edge_val_kernel(
    const int* __restrict__ srcIdx, const int* __restrict__ dstIdx,
    const float* __restrict__ rbf, const float* __restrict__ rsh,
    const float* __restrict__ sd, const float* __restrict__ Wrbf,
    const f16* __restrict__ PB, const float* __restrict__ aexp,
    const float* __restrict__ segsum, float* __restrict__ msg) {
  __shared__ EdgeSmem S;
  int w = threadIdx.x >> 5, lane = threadIdx.x & 31;
  int tile = blockIdx.x * NW + w;
  build_tiles(tile, w, lane, srcIdx, dstIdx, rbf, rsh, sd, Wrbf, S);
  for (int i = lane; i < 128; i += 32) {
    int m = i >> 3, h = i & 7;
    S.lacc[w][m][h] = aexp[(size_t)(tile * 16 + m) * 8 + h] /
                      (segsum[(size_t)S.dstI[w][m] * 8 + h] + 1e-16f);
  }
  __syncthreads();
  int half = lane >> 4, n = lane & 15;

  v16h A0k0 = load_A(&S.A[w][0][0][0], 0, lane);
  v16h A0k1 = load_A(&S.A[w][0][0][0], 32, lane);
  v16h A1k0[3], A1k1[3];
#pragma unroll
  for (int t = 0; t < 3; ++t) {
    A1k0[t] = load_A(&S.A[w][1 + t][0][0], 0, lane);
    A1k1[t] = load_A(&S.A[w][1 + t][0][0], 32, lane);
  }
  const f16* PBv0 = PB + 16384;
  const f16* PBv1 = PB + 16384 * 2 + 8192;

  for (int h = 0; h < 8; ++h) {
    // kv0 val n-tiles nb = 2h, 2h+1
#pragma unroll
    for (int s = 0; s < 2; ++s) {
      int nb = 2 * h + s;
      v8f acc = {};
      acc = wmma_f16(A0k0, load_B(PBv0, nb * 2 + 0, lane), acc);
      acc = wmma_f16(A0k1, load_B(PBv0, nb * 2 + 1, lane), acc);
      int j = s * 16 + n;
#pragma unroll
      for (int r = 0; r < 8; ++r) {
        int m = half * 8 + r;
        atomicAdd(&msg[(size_t)S.dstI[w][m] * 640 + h * 80 + j],
                  S.lacc[w][m][h] * acc[r]);
      }
    }
    // kv1 val comps t = 0..2, n-tile nb = h
#pragma unroll
    for (int t = 0; t < 3; ++t) {
      v8f acc = {};
      acc = wmma_f16(A1k0[t], load_B(PBv1, h * 2 + 0, lane), acc);
      acc = wmma_f16(A1k1[t], load_B(PBv1, h * 2 + 1, lane), acc);
      int j = 32 + n * 3 + t;
#pragma unroll
      for (int r = 0; r < 8; ++r) {
        int m = half * 8 + r;
        atomicAdd(&msg[(size_t)S.dstI[w][m] * 640 + h * 80 + j],
                  S.lacc[w][m][h] * acc[r]);
      }
    }
  }
}

// ------------------------------------------------------------- out pass -----
__global__ __launch_bounds__(256) void out_kernel(
    const float* __restrict__ node, const float* __restrict__ msg,
    const float* __restrict__ Wmsg0, const float* __restrict__ Wmsg1,
    float* __restrict__ out) {
  int w = threadIdx.x >> 5, lane = threadIdx.x & 31;
  int nid = blockIdx.x * 8 + w;
  const float* m = msg + (size_t)nid * 640;
  const float* nr = node + (size_t)nid * 160;
  float* o = out + (size_t)nid * 160;
  for (int c = lane; c < 160; c += 32) {
    float acc = 0.f;
    if (c < 64) {
      for (int i = 0; i < 256; ++i) {
        int h = i >> 5, jj = i & 31;
        acc += m[h * 80 + jj] * Wmsg0[i * 64 + c];
      }
      acc *= 0.0625f;
    } else {
      int jj = c - 64, v = jj / 3, t = jj % 3;
      for (int u = 0; u < 128; ++u) {
        int h = u >> 4, uu = u & 15;
        acc += m[h * 80 + 32 + uu * 3 + t] * Wmsg1[u * 32 + v];
      }
      acc *= INV_SQRT128;
    }
    o[c] = nr[c] + acc;
  }
}

// ---------------------------------------------------------------- launch ----
extern "C" void kernel_launch(void* const* d_in, const int* in_sizes, int n_in,
                              void* d_out, int out_size, void* d_ws,
                              size_t ws_size, hipStream_t stream) {
  const float* node  = (const float*)d_in[0];
  const float* rbf   = (const float*)d_in[1];
  const float* rsh   = (const float*)d_in[2];
  const int*   eidx  = (const int*)d_in[3];
  const float* gamma0= (const float*)d_in[4];
  const float* gamma1= (const float*)d_in[5];
  const float* Wq0   = (const float*)d_in[6];
  const float* Wq1   = (const float*)d_in[7];
  const float* Wsrc0 = (const float*)d_in[8];
  const float* Wsrc1 = (const float*)d_in[9];
  const float* Wdst0 = (const float*)d_in[10];
  const float* Wdst1 = (const float*)d_in[11];
  const float* Wrbf  = (const float*)d_in[12];
  const float* Wkv0  = (const float*)d_in[13];
  const float* Wkv1  = (const float*)d_in[14];
  const float* Wmsg0 = (const float*)d_in[15];
  const float* Wmsg1 = (const float*)d_in[16];
  float* out = (float*)d_out;

  float* q      = (float*)d_ws;                  // NN*640
  float* sd     = q + (size_t)NN * 640;          // NN*160
  float* logits = sd + (size_t)NN * 160;         // NE*8
  float* segmax = logits + (size_t)NE * 8;       // NN*8
  float* segsum = segmax + (size_t)NN * 8;       // NN*8
  float* msg    = segsum + (size_t)NN * 8;       // NN*640
  f16*   PB     = (f16*)(msg + (size_t)NN * 640);// 49152 f16

  const int* srcI = eidx;
  const int* dstI = eidx + NE;

  prepack_kernel<<<192, 256, 0, stream>>>(Wkv0, Wkv1, PB);
  init_kernel<<<(NN * 640 + 255) / 256, 256, 0, stream>>>(segmax, segsum, msg);
  node_kernel<<<NN / 8, 256, 0, stream>>>(node, gamma0, gamma1, Wq0, Wq1,
                                          Wsrc0, Wsrc1, Wdst0, Wdst1, q, sd);
  edge_key_kernel<<<NTILES / NW, 64, 0, stream>>>(srcI, dstI, rbf, rsh, sd,
                                                  Wrbf, PB, q, logits, segmax);
  softmax_kernel<<<(NE * 8 + 255) / 256, 256, 0, stream>>>(dstI, segmax,
                                                           logits, segsum);
  edge_val_kernel<<<NTILES / NW, 64, 0, stream>>>(srcI, dstI, rbf, rsh, sd,
                                                  Wrbf, PB, logits, segsum, msg);
  out_kernel<<<NN / 8, 256, 0, stream>>>(node, msg, Wmsg0, Wmsg1, out);
}
